// ChannelSELayerOwn_80066780332137
// MI455X (gfx1250) — matrix operand order repaired and checked by reading, not compiled, not run on hardware
//
#include <hip/hip_runtime.h>
#include <hip/hip_bf16.h>

typedef __attribute__((ext_vector_type(16))) _Float16 v16h;
typedef __attribute__((ext_vector_type(8)))  float    v8f;

#define C_CH   64
#define B_N    8
#define R_K    8
#define SPAT   110592           // 48*48*48
#define SPAT4  27648            // SPAT/4

// ---------------------------------------------------------------------------
// Kernel 1: global average pool.  One block per (b,c) channel.
// 226 MB read @ 23.3 TB/s  ->  ~10 us.  float4 loads, 4 accumulators in
// flight, deterministic LDS tree reduction.
// ---------------------------------------------------------------------------
__global__ __launch_bounds__(256)
void se_pool_kernel(const float4* __restrict__ x4, float* __restrict__ means) {
    __shared__ float red[256];
    const int bc  = blockIdx.x;                 // 0..511 = b*64+c
    const int tid = threadIdx.x;
    const float4* p = x4 + (long)bc * SPAT4;

    float s0 = 0.f, s1 = 0.f, s2 = 0.f, s3 = 0.f;
    // 27648 = 27 * 1024 ; 4 independent 16B loads per iteration
    #pragma unroll 3
    for (int it = 0; it < 27; ++it) {
        const int base = it * 1024 + tid;
        float4 v0 = p[base];
        float4 v1 = p[base + 256];
        float4 v2 = p[base + 512];
        float4 v3 = p[base + 768];
        s0 += (v0.x + v0.y) + (v0.z + v0.w);
        s1 += (v1.x + v1.y) + (v1.z + v1.w);
        s2 += (v2.x + v2.y) + (v2.z + v2.w);
        s3 += (v3.x + v3.y) + (v3.z + v3.w);
    }
    red[tid] = (s0 + s1) + (s2 + s3);
    __syncthreads();
    for (int w = 128; w > 0; w >>= 1) {
        if (tid < w) red[tid] += red[tid + w];
        __syncthreads();
    }
    if (tid == 0) means[bc] = red[0] * (1.0f / (float)SPAT);
}

// ---------------------------------------------------------------------------
// Kernel 2: fused MLP (two 16x64 @ 64x64 GEMMs via v_wmma_f32_16x16x32_f16)
// + per-row top-8.  Single wave32 block; EXEC is all ones at every WMMA.
//
// Fragment layouts (ISA 7.12.2, wave32):
//   A 16x32 f16 : lane L -> M = L&15 ;  elem e -> K = (e>>3)*16 + ((L>>4)&1)*8 + (e&7)
//   B 32x16 f16 : lane L -> N = L&15 ;  elem e -> K = (e>>3)*16 + ((L>>4)&1)*8 + (e&7)
//   C 16x16 f32 : lane L -> N = L&15 ;  elem r -> M = r + ((L>>4)&1)*8
// ---------------------------------------------------------------------------
__global__ __launch_bounds__(32)
void se_mlp_topk_kernel(const float* __restrict__ means,
                        const float* __restrict__ w1, const float* __restrict__ b1,
                        const float* __restrict__ w2, const float* __restrict__ b2,
                        int* __restrict__ idx_out) {
    __shared__ float ybuf[16 * 64];

    const int lane = threadIdx.x;       // 0..31
    const int m    = lane & 15;         // A-row / C-N column index
    const int half = (lane >> 4) & 1;   // which 16-lane half

    for (int layer = 0; layer < 2; ++layer) {
        const float* W = layer ? w2 : w1;
        const float* Bv = layer ? b2 : b1;

        // ---- build A fragments (two K-steps of 32) from pooled means / ybuf
        v16h a0, a1;
        const float rowScale = (m < 8) ? 1.0f : 0.0f;   // zero-pad rows 8..15 (no branch)
        const int   mrow     = m & 7;
        #pragma unroll
        for (int e = 0; e < 16; ++e) {
            const int kl = ((e >> 3) << 4) + (half << 3) + (e & 7);  // 0..31
            float v0, v1;
            if (layer == 0) {
                v0 = means[mrow * 64 + kl]      * rowScale;
                v1 = means[mrow * 64 + 32 + kl] * rowScale;
            } else {
                v0 = ybuf[m * 64 + kl];
                v1 = ybuf[m * 64 + 32 + kl];
            }
            a0[e] = (_Float16)v0;
            a1[e] = (_Float16)v1;
        }
        __syncthreads();   // all ybuf reads done before this layer's writes

        // ---- 4 N-tiles of 16 output columns each
        #pragma unroll
        for (int t = 0; t < 4; ++t) {
            const int n = m;                         // output column within tile
            const float bias = Bv[t * 16 + n];
            v8f acc = { bias, bias, bias, bias, bias, bias, bias, bias };

            v16h bf0, bf1;                           // B[k][n] = W[(t*16+n)*64 + k]
            #pragma unroll
            for (int e = 0; e < 16; ++e) {
                const int k0 = ((e >> 3) << 4) + (half << 3) + (e & 7);        // K-step 0: k in [0,32)
                bf0[e] = (_Float16)W[(t * 16 + n) * 64 + k0];
                bf1[e] = (_Float16)W[(t * 16 + n) * 64 + 32 + k0];
            }

            acc = __builtin_amdgcn_wmma_f32_16x16x32_f16(false, a0, false, bf0,
                                                         (short)0, acc, false, false);
            acc = __builtin_amdgcn_wmma_f32_16x16x32_f16(false, a1, false, bf1,
                                                         (short)0, acc, false, false);

            // ---- activation + write back to LDS
            #pragma unroll
            for (int r = 0; r < 8; ++r) {
                float v = acc[r];
                if (layer == 0) v = (v > 0.f) ? v : 0.01f * v;           // LeakyReLU
                else            v = 1.0f / (1.0f + __expf(-v));          // Sigmoid
                const int Mrow = r + half * 8;
                ybuf[Mrow * 64 + t * 16 + n] = v;
            }
        }
        __syncthreads();
    }

    // ---- top-8 per batch row (lanes 0..7), ties -> lowest index, descending
    if (lane < 8) {
        #pragma unroll
        for (int r = 0; r < R_K; ++r) {
            float best = -1e30f; int bi = 0;
            for (int j = 0; j < 64; ++j) {
                const float v = ybuf[lane * 64 + j];
                if (v > best) { best = v; bi = j; }
            }
            ybuf[lane * 64 + bi] = -1e30f;
            idx_out[lane * R_K + r] = bi;
        }
    }
}

// ---------------------------------------------------------------------------
// Kernel 3: channel gather.  grid (108, 64): grid.y = b*8+r, 108*256 threads
// copy the 27648 float4 of one selected channel.  ~57 MB -> ~2.5 us.
// ---------------------------------------------------------------------------
__global__ __launch_bounds__(256)
void se_gather_kernel(const float4* __restrict__ x4, const int* __restrict__ idx,
                      float4* __restrict__ out4) {
    const int pair = blockIdx.y;            // b*8 + r
    const int b    = pair >> 3;
    const int c    = idx[pair];
    const long src = (long)(b * C_CH + c) * SPAT4;
    const long dst = (long)pair * SPAT4;
    const int  i   = blockIdx.x * 256 + threadIdx.x;   // 0..27647
    out4[dst + i] = x4[src + i];
}

// ---------------------------------------------------------------------------
extern "C" void kernel_launch(void* const* d_in, const int* in_sizes, int n_in,
                              void* d_out, int out_size, void* d_ws, size_t ws_size,
                              hipStream_t stream) {
    const float* x  = (const float*)d_in[0];
    const float* w1 = (const float*)d_in[1];
    const float* b1 = (const float*)d_in[2];
    const float* w2 = (const float*)d_in[3];
    const float* b2 = (const float*)d_in[4];

    float* means = (float*)d_ws;             // 512 floats
    int*   idxp  = (int*)d_ws + 512;         // 64 ints

    se_pool_kernel<<<B_N * C_CH, 256, 0, stream>>>((const float4*)x, means);
    se_mlp_topk_kernel<<<1, 32, 0, stream>>>(means, w1, b1, w2, b2, idxp);

    dim3 ggrid(SPAT4 / 256, B_N * R_K);
    se_gather_kernel<<<ggrid, 256, 0, stream>>>((const float4*)x, idxp, (float4*)d_out);
}
